// T5SelfAttention_79302276153517
// MI455X (gfx1250) — matrix operand order repaired and checked by reading, not compiled, not run on hardware
//
#include <hip/hip_runtime.h>

typedef __attribute__((ext_vector_type(16))) __bf16 v16bf;
typedef __attribute__((ext_vector_type(8)))  float  v8f;
typedef __attribute__((ext_vector_type(4)))  unsigned int u32x4;
typedef __attribute__((ext_vector_type(4)))  int i32x4;
typedef unsigned int  u32;
typedef unsigned short u16;

union Frag { v16bf v; u32 u[8]; u32x4 q[2]; };

#if defined(__has_builtin)
#  if __has_builtin(__builtin_amdgcn_global_load_async_to_lds_b128) && \
      __has_builtin(__builtin_amdgcn_s_wait_asynccnt)
#    define HAVE_ASYNC_COPY 1
#  endif
#endif
#ifndef HAVE_ASYNC_COPY
#  define HAVE_ASYNC_COPY 0
#endif
#define AS1 __attribute__((address_space(1)))
#define AS3 __attribute__((address_space(3)))

__device__ __forceinline__ u16 f32_bf16(float f) {
  u32 u = __float_as_uint(f);
  u += 0x7FFFu + ((u >> 16) & 1u);  // round-to-nearest-even
  return (u16)(u >> 16);
}

// ---------------------------------------------------------------------------
// One-time conversions
// ---------------------------------------------------------------------------
__global__ __launch_bounds__(256)
void cvt_bf16(const float* __restrict__ in, u16* __restrict__ out, int n2) {
  int i = blockIdx.x * 256 + threadIdx.x;  // handles 2 elements
  if (i < n2) {
    u32 p = (u32)f32_bf16(in[2 * i]) | ((u32)f32_bf16(in[2 * i + 1]) << 16);
    ((u32*)out)[i] = p;
  }
}

// W[k][n] f32 (1024x1024)  ->  WT[n][k] bf16
__global__ __launch_bounds__(256)
void cvt_w_t(const float* __restrict__ W, u16* __restrict__ WT) {
  __shared__ u16 tile[32 * 33];
  const int k0 = blockIdx.y * 32, n0 = blockIdx.x * 32;
  const int t = threadIdx.x;
#pragma unroll
  for (int i = 0; i < 4; i++) {
    int idx = t + i * 256, r = idx >> 5, c = idx & 31;  // r=k, c=n
    tile[c * 33 + r] = f32_bf16(W[(size_t)(k0 + r) * 1024 + n0 + c]);
  }
  __syncthreads();
#pragma unroll
  for (int i = 0; i < 2; i++) {
    int idx = t + i * 256, n = idx >> 4, kp = idx & 15;  // 512 dwords
    u32 p = (u32)tile[n * 33 + 2 * kp] | ((u32)tile[n * 33 + 2 * kp + 1] << 16);
    ((u32*)WT)[(size_t)(n0 + n) * 512 + (k0 >> 1) + kp] = p;
  }
}

// ---------------------------------------------------------------------------
// GEMM: C[4096 x 1024] = A_bf16[4096 x 1024](row-major) * WT_bf16[1024 x 1024]
// (WT is pre-transposed: [n][k]).
// OUTMODE 0: bf16 scattered to [b][h][i][d], scaled.  OUTMODE 1: f32 row-major.
// ---------------------------------------------------------------------------
template <int OUTMODE>
__global__ __launch_bounds__(256)
void gemm128(const u16* __restrict__ A, const u16* __restrict__ WT,
             void* __restrict__ outp, float scale) {
  constexpr int K = 1024, NC = 1024;
  // row stride 40 u16 = 80 B (16-aligned, odd bank-group padding)
  __shared__ __align__(16) u16 Xs[128 * 40];
  __shared__ __align__(16) u16 Ws[128 * 40];

  const int t = threadIdx.x;
  const int lane = t & 31, wave = t >> 5;
  const int wm = wave >> 2, wn = wave & 3;  // 2x4 wave grid
  const int hi = lane >> 4, lo = lane & 15;
  const int m0 = blockIdx.y * 128, n0 = blockIdx.x * 128;

  const v8f vz = {0.f, 0.f, 0.f, 0.f, 0.f, 0.f, 0.f, 0.f};
  v8f acc[4][2];
#pragma unroll
  for (int mi = 0; mi < 4; mi++)
#pragma unroll
    for (int ni = 0; ni < 2; ni++) acc[mi][ni] = vz;

  const u32* Ag = (const u32*)A;    // 512 dwords per row
  const u32* Wg = (const u32*)WT;   // 512 dwords per row

  for (int k0 = 0; k0 < K; k0 += 32) {
#if HAVE_ASYNC_COPY
    // per-lane 16B async copies: 512 chunks per tile, 2 per thread per tile
#pragma unroll
    for (int i = 0; i < 2; i++) {
      int idx = t + i * 256;                 // 0..511
      int r = idx >> 2, c = idx & 3;         // row, 16B-chunk in row
      const char* gx = (const char*)&Ag[(size_t)(m0 + r) * 512 + (k0 >> 1)] + c * 16;
      const char* gw = (const char*)&Wg[(size_t)(n0 + r) * 512 + (k0 >> 1)] + c * 16;
      char* lx = (char*)Xs + r * 80 + c * 16;
      char* lw = (char*)Ws + r * 80 + c * 16;
      __builtin_amdgcn_global_load_async_to_lds_b128((AS1 i32x4*)gx, (AS3 i32x4*)lx, 0, 0);
      __builtin_amdgcn_global_load_async_to_lds_b128((AS1 i32x4*)gw, (AS3 i32x4*)lw, 0, 0);
    }
    __builtin_amdgcn_s_wait_asynccnt(0);
#else
    u32* XsU = (u32*)Xs;
    u32* WsU = (u32*)Ws;
#pragma unroll
    for (int i = 0; i < 8; i++) {
      int idx = t + i * 256;                 // 0..2047
      int r = idx >> 4, c = idx & 15;        // row, dword in k-run
      XsU[r * 20 + c] = Ag[(size_t)(m0 + r) * 512 + (k0 >> 1) + c];
      WsU[r * 20 + c] = Wg[(size_t)(n0 + r) * 512 + (k0 >> 1) + c];
    }
#endif
    if (k0 + 32 < K) {  // hint next tiles toward the caches
      __builtin_prefetch(&Ag[(size_t)(m0 + (t >> 1)) * 512 + ((k0 + 32) >> 1)], 0, 1);
      __builtin_prefetch(&Wg[(size_t)(n0 + (t >> 1)) * 512 + ((k0 + 32) >> 1)], 0, 1);
    }
    __syncthreads();

    Frag a[4], b[2];
#pragma unroll
    for (int mi = 0; mi < 4; mi++) {
      int row = wm * 64 + mi * 16 + lo;
      a[mi].q[0] = *(const u32x4*)&Xs[row * 40 + hi * 8];        // K = hi*8+0..7
      a[mi].q[1] = *(const u32x4*)&Xs[row * 40 + 16 + hi * 8];   // K = 16+hi*8+0..7
    }
#pragma unroll
    for (int ni = 0; ni < 2; ni++) {
      int col = wn * 32 + ni * 16 + lo;
      b[ni].q[0] = *(const u32x4*)&Ws[col * 40 + hi * 8];        // K = hi*8+0..7
      b[ni].q[1] = *(const u32x4*)&Ws[col * 40 + 16 + hi * 8];   // K = 16+hi*8+0..7
    }
#pragma unroll
    for (int mi = 0; mi < 4; mi++)
#pragma unroll
      for (int ni = 0; ni < 2; ni++)
        acc[mi][ni] = __builtin_amdgcn_wmma_f32_16x16x32_bf16(
            false, a[mi].v, false, b[ni].v, (short)0, acc[mi][ni], false, false);
    __syncthreads();
  }

#pragma unroll
  for (int mi = 0; mi < 4; mi++)
#pragma unroll
    for (int ni = 0; ni < 2; ni++)
#pragma unroll
      for (int j = 0; j < 8; j++) {
        int gm = m0 + wm * 64 + mi * 16 + hi * 8 + j;
        int gn = n0 + wn * 32 + ni * 16 + lo;
        float val = acc[mi][ni][j] * scale;
        if (OUTMODE == 0) {
          int bb = gm >> 11, ii = gm & 2047, hh = gn >> 6, dd = gn & 63;
          ((u16*)outp)[(((size_t)bb * 16 + hh) * 2048 + ii) * 64 + dd] = f32_bf16(val);
        } else {
          ((float*)outp)[(size_t)gm * NC + gn] = val;
        }
      }
}

// ---------------------------------------------------------------------------
// Flash attention (causal): block = (b, h, 64-row tile), 4 waves.
// Output bf16 [b][i][h*64+d] (row-major A for the final GEMM).
// ---------------------------------------------------------------------------
__global__ __launch_bounds__(128)
void flash_attn(const u16* __restrict__ Qb, const u16* __restrict__ Kb,
                const u16* __restrict__ Vb, u16* __restrict__ Ao) {
  __shared__ __align__(16) u16 Ks[64 * 64];     // [key][d]
  __shared__ __align__(16) u16 Vt[64 * 72];     // [d][key], 144B stride
  __shared__ __align__(16) u16 Ps[4][16 * 64];  // wave-private P

  const int t = threadIdx.x, lane = t & 31, wave = t >> 5;
  const int hi = lane >> 4, lo = lane & 15;
  const int rb = blockIdx.x, h = blockIdx.y, b = blockIdx.z;
  const int bh = b * 16 + h;
  const u16* Qh = Qb + (size_t)bh * 2048 * 64;
  const u16* Kh = Kb + (size_t)bh * 2048 * 64;
  const u16* Vh = Vb + (size_t)bh * 2048 * 64;
  const int i0 = rb * 64 + wave * 16;

  // Q fragments (A-layout), direct b128 global loads
  Frag aq[2];
#pragma unroll
  for (int dc = 0; dc < 2; dc++) {
    const u16* qp = Qh + (size_t)(i0 + lo) * 64 + dc * 32;
    aq[dc].q[0] = *(const u32x4*)(qp + hi * 8);
    aq[dc].q[1] = *(const u32x4*)(qp + 16 + hi * 8);
  }

  const v8f vz = {0.f, 0.f, 0.f, 0.f, 0.f, 0.f, 0.f, 0.f};
  v8f acco[4];
#pragma unroll
  for (int ds = 0; ds < 4; ds++) acco[ds] = vz;
  float mrow[8], lrow[8];
#pragma unroll
  for (int j = 0; j < 8; j++) { mrow[j] = -3.0e38f; lrow[j] = 0.f; }

  for (int jb = 0; jb <= rb; ++jb) {
    __syncthreads();
    const u32* Kg = (const u32*)(Kh + (size_t)jb * 64 * 64);
    const u32* Vg = (const u32*)(Vh + (size_t)jb * 64 * 64);
#if HAVE_ASYNC_COPY
#pragma unroll
    for (int i = 0; i < 4; i++) {  // 512 x 16B chunks, linear
      int off = (t + i * 128) * 16;
      __builtin_amdgcn_global_load_async_to_lds_b128(
          (AS1 i32x4*)((const char*)Kg + off), (AS3 i32x4*)((char*)Ks + off), 0, 0);
    }
#else
    u32* KsU = (u32*)Ks;
#pragma unroll
    for (int i = 0; i < 16; i++) KsU[t + i * 128] = Kg[t + i * 128];
#endif
    // V transposed into LDS: read [key][d] dwords, scatter [d][key] u16
#pragma unroll
    for (int i = 0; i < 16; i++) {
      int idx = t + i * 128;                 // 0..2047 dwords
      int key = idx >> 5, dp = (idx & 31) * 2;
      u32 v = Vg[idx];
      Vt[dp * 72 + key] = (u16)v;
      Vt[(dp + 1) * 72 + key] = (u16)(v >> 16);
    }
#if HAVE_ASYNC_COPY
    __builtin_amdgcn_s_wait_asynccnt(0);
#endif
    __syncthreads();

    // S = Q * K^T
    v8f accs[4];
#pragma unroll
    for (int sn = 0; sn < 4; sn++) {
      v8f s = vz;
#pragma unroll
      for (int dc = 0; dc < 2; dc++) {
        Frag bk;
        const u16* kp = &Ks[(sn * 16 + lo) * 64 + dc * 32 + hi * 8];
        bk.q[0] = *(const u32x4*)kp;
        bk.q[1] = *(const u32x4*)(kp + 16);
        s = __builtin_amdgcn_wmma_f32_16x16x32_bf16(false, aq[dc].v, false, bk.v,
                                                    (short)0, s, false, false);
      }
      accs[sn] = s;
    }

    if (jb == rb) {  // causal mask, diagonal block only
#pragma unroll
      for (int sn = 0; sn < 4; sn++)
#pragma unroll
        for (int j = 0; j < 8; j++)
          if (jb * 64 + sn * 16 + lo > i0 + hi * 8 + j) accs[sn][j] = -3.0e38f;
    }

    // online softmax
    float al[8];
#pragma unroll
    for (int j = 0; j < 8; j++) {
      float mx = fmaxf(fmaxf(accs[0][j], accs[1][j]), fmaxf(accs[2][j], accs[3][j]));
      mx = fmaxf(mx, __shfl_xor(mx, 1, 32));
      mx = fmaxf(mx, __shfl_xor(mx, 2, 32));
      mx = fmaxf(mx, __shfl_xor(mx, 4, 32));
      mx = fmaxf(mx, __shfl_xor(mx, 8, 32));
      float mn = fmaxf(mrow[j], mx);
      al[j] = __expf(mrow[j] - mn);
      mrow[j] = mn;
      float pj = 0.f;
#pragma unroll
      for (int sn = 0; sn < 4; sn++) {
        float p = __expf(accs[sn][j] - mn);
        pj += p;
        Ps[wave][(hi * 8 + j) * 64 + sn * 16 + lo] = f32_bf16(p);
      }
      pj += __shfl_xor(pj, 1, 32);
      pj += __shfl_xor(pj, 2, 32);
      pj += __shfl_xor(pj, 4, 32);
      pj += __shfl_xor(pj, 8, 32);
      lrow[j] = lrow[j] * al[j] + pj;
    }
#pragma unroll
    for (int ds = 0; ds < 4; ds++)
#pragma unroll
      for (int j = 0; j < 8; j++) acco[ds][j] *= al[j];

    // O += P * V  (P via LDS layout swap, V from transposed tile: all b128)
    Frag ap[2];
#pragma unroll
    for (int kc = 0; kc < 2; kc++) {
      const u16* pp = &Ps[wave][lo * 64 + kc * 32];
      ap[kc].q[0] = *(const u32x4*)(pp + hi * 8);
      ap[kc].q[1] = *(const u32x4*)(pp + 16 + hi * 8);
    }
#pragma unroll
    for (int ds = 0; ds < 4; ds++) {
      v8f o = acco[ds];
#pragma unroll
      for (int kc = 0; kc < 2; kc++) {
        Frag bv;
        const u16* vp = &Vt[(ds * 16 + lo) * 72 + kc * 32 + hi * 8];
        bv.q[0] = *(const u32x4*)vp;
        bv.q[1] = *(const u32x4*)(vp + 16);
        o = __builtin_amdgcn_wmma_f32_16x16x32_bf16(false, ap[kc].v, false, bv.v,
                                                    (short)0, o, false, false);
      }
      acco[ds] = o;
    }
  }

  // normalize, write bf16 [b][i][h*64+d]
#pragma unroll
  for (int j = 0; j < 8; j++) {
    float inv = 1.0f / lrow[j];
    int ig = i0 + hi * 8 + j;
#pragma unroll
    for (int ds = 0; ds < 4; ds++)
      Ao[((size_t)b * 2048 + ig) * 1024 + h * 64 + ds * 16 + lo] =
          f32_bf16(acco[ds][j] * inv);
  }
}

// ---------------------------------------------------------------------------
extern "C" void kernel_launch(void* const* d_in, const int* in_sizes, int n_in,
                              void* d_out, int out_size, void* d_ws, size_t ws_size,
                              hipStream_t stream) {
  const float* x  = (const float*)d_in[0];
  const float* Wq = (const float*)d_in[1];
  const float* Wk = (const float*)d_in[2];
  const float* Wv = (const float*)d_in[3];
  const float* Wo = (const float*)d_in[4];

  const size_t xe = (size_t)2 * 2048 * 1024;  // 4,194,304
  const size_t we = (size_t)1024 * 1024;      // 1,048,576
  u16* xbf  = (u16*)d_ws;
  u16* wqT  = xbf + xe;
  u16* wkT  = wqT + we;
  u16* wvT  = wkT + we;
  u16* woT  = wvT + we;
  u16* Qbf  = woT + we;
  u16* Kbf  = Qbf + xe;
  u16* Vbf  = Kbf + xe;
  u16* aobf = Vbf + xe;   // total ~48 MiB

  cvt_bf16<<<(int)(xe / 2 / 256), 256, 0, stream>>>(x, xbf, (int)(xe / 2));
  dim3 gw(32, 32);
  cvt_w_t<<<gw, 256, 0, stream>>>(Wq, wqT);
  cvt_w_t<<<gw, 256, 0, stream>>>(Wk, wkT);
  cvt_w_t<<<gw, 256, 0, stream>>>(Wv, wvT);
  cvt_w_t<<<gw, 256, 0, stream>>>(Wo, woT);

  dim3 gg(1024 / 128, 4096 / 128);  // (8, 32)
  gemm128<0><<<gg, 256, 0, stream>>>(xbf, wqT, (void*)Qbf, 0.125f);  // 1/sqrt(64)
  gemm128<0><<<gg, 256, 0, stream>>>(xbf, wkT, (void*)Kbf, 1.0f);
  gemm128<0><<<gg, 256, 0, stream>>>(xbf, wvT, (void*)Vbf, 1.0f);
  flash_attn<<<dim3(2048 / 64, 16, 2), 128, 0, stream>>>(Qbf, Kbf, Vbf, aobf);
  gemm128<1><<<gg, 256, 0, stream>>>(aobf, woT, d_out, 1.0f);
}
